// ConvLRULayer_326417514518
// MI455X (gfx1250) — compile-verified, matrix operand order checked by reading, not run on hardware
//
#include <hip/hip_runtime.h>

// ---------------------------------------------------------------------------
// ConvLRU layer for MI455X (gfx1250, wave32, WMMA + async-to-LDS).
//
//  * f32 WMMA (V_WMMA_F32_16X16X4_F32) for low-rank proj/recon, channel
//    projection, 1x1 fuse: exact fp32; these stages are memory-bound
//    (layer floor ~20us at 23.3 TB/s) so the f32 matrix rate is free.
//  * bf16 WMMA (V_WMMA_F32_16X16X32_BF16) for the two 3x3 convs (~38 GFLOP,
//    the only compute-limited stage: VALU ~400us, bf16 WMMA ~10us).
//  * GLOBAL_LOAD_ASYNC_TO_LDS_B128 (+ s_wait_asynccnt) for all bulk
//    global->LDS staging (V tiles, weight blocks, conv halo slab).
//  * Padded conv input stored channel-innermost so bf16 B fragments are two
//    ds_load_b128 per lane and the halo slab is one contiguous async region.
//
// Workspace ~206 MB; all launches stream-ordered (graph-capture safe).
// ---------------------------------------------------------------------------

#define DEV __device__ __forceinline__

constexpr int B_  = 2,  L_ = 16, C_ = 64, S_ = 64, W_ = 128, R_ = 32, MH_ = 32;
constexpr int BL_ = B_ * L_;      // 32
constexpr int SW_ = S_ * W_;      // 8192
constexpr int CR_ = C_ * R_;      // 2048
constexpr int SP_ = S_ + 2;       // 66  (padded rows)
constexpr int WP_ = W_ + 2;       // 130 (padded cols)

typedef float  v2f   __attribute__((ext_vector_type(2)));
typedef float  v8f   __attribute__((ext_vector_type(8)));
typedef __bf16 v8bf  __attribute__((ext_vector_type(8)));
typedef __bf16 v16bf __attribute__((ext_vector_type(16)));

// ------------------------------ WMMA wrappers ------------------------------
DEV v8f wmma4(v2f a, v2f b, v8f c) {
  // D = A(16x4 f32) * B(4x16 f32) + C(16x16 f32)
  return __builtin_amdgcn_wmma_f32_16x16x4_f32(false, a, false, b, (short)0, c,
                                               false, false);
}
DEV v8f wmma32bf(v16bf a, v16bf b, v8f c) {
  // D = A(16x32 bf16) * B(32x16 bf16) + C(16x16 f32)
  return __builtin_amdgcn_wmma_f32_16x16x32_bf16(false, a, false, b, (short)0,
                                                 c, false, false);
}

// ----------------------- async global->LDS staging -------------------------
// GLOBAL_LOAD_ASYNC_TO_LDS_B128: VDST = LDS byte offset VGPR, VADDR = 64-bit
// global address.  Tracked with ASYNCcnt (CDNA5 ISA).
DEV unsigned lds_off_of(const void* p) {
  return (unsigned)(unsigned long long)(uintptr_t)
         (__attribute__((address_space(3))) const char*)p;
}
DEV void async_b128(unsigned lds_off, const void* gptr) {
  asm volatile("global_load_async_to_lds_b128 %0, %1, off"
               :: "v"(lds_off), "v"((unsigned long long)(uintptr_t)gptr)
               : "memory");
}
DEV void wait_async() {
  asm volatile("s_wait_asynccnt 0x0" ::: "memory");
}
// Cooperative contiguous copy (bytes multiple of 16), one b128 per lane/iter.
DEV void stage_async(void* lds, const void* g, unsigned bytes) {
  unsigned base = lds_off_of(lds);
  const char* gp = (const char*)g;
  for (unsigned t = threadIdx.x * 16u; t < bytes; t += 256u * 16u)
    async_b128(base + t, gp + t);
}

// ---------------------- fragment loaders / C store -------------------------
// f32 A 16x4 (MxK), row-major src at [m0][k0], ld = row stride.
DEV v2f ldA4(const float* p, int ld) {
  int lane = threadIdx.x & 31;
  int m = lane & 15, kk = (lane >> 4) << 1;
  const float* q = p + m * ld + kk;
  return v2f{q[0], q[1]};
}
// f32 B 4x16 (KxN), row-major src at [k0][n0], ld = row stride.
DEV v2f ldB4(const float* p, int ld) {
  int lane = threadIdx.x & 31;
  int n = lane & 15, kk = (lane >> 4) << 1;
  return v2f{p[kk * ld + n], p[(kk + 1) * ld + n]};
}
// f32 B 4x16 where B[k][n] = V[n0+n][k0+k], V row-major (W_ x R_).
DEV v2f ldB4T(const float* v, int n0, int k0) {
  int lane = threadIdx.x & 31;
  int n = lane & 15, kk = (lane >> 4) << 1;
  const float* p = v + (n0 + n) * R_ + k0 + kk;
  return v2f{p[0], p[1]};
}
// C/D 16x16 f32 store: VGPR i -> M=i (lanes 0-15) / M=8+i (lanes 16-31).
DEV void stC(float* dst, int ld, v8f c) {
  int lane = threadIdx.x & 31;
  int n = lane & 15, rb = (lane >> 4) * 8;
#pragma unroll
  for (int i = 0; i < 8; ++i) dst[(rb + i) * ld + n] = c[i];
}
// bf16 A 16x32 (MxK), K innermost: element e -> K = e + (e>=8?8:0) + khi,
// i.e. two contiguous 8-element (16B) runs at khi and khi+16.
DEV v16bf ldA16v(const __bf16* p, int ld) {
  int lane = threadIdx.x & 31;
  int m = lane & 15, khi = (lane >> 4) * 8;
  const __bf16* q = p + m * ld + khi;
  v8bf lo = *(const v8bf*)q;
  v8bf hi = *(const v8bf*)(q + 16);
  return __builtin_shufflevector(lo, hi, 0, 1, 2, 3, 4, 5, 6, 7,
                                 8, 9, 10, 11, 12, 13, 14, 15);
}
// bf16 B 32x16 from channel-innermost halo slab [dy][w][ch]:
// B[k][n] = slab[(dy*WP_ + wbase + n)*64 + cc*32 + k]; two 16B LDS loads.
DEV v16bf ldB16slab(const __bf16* slab, int cc, int dy, int wbase) {
  int lane = threadIdx.x & 31;
  int n = lane & 15, khi = (lane >> 4) * 8;
  const __bf16* p = slab + ((dy * WP_ + wbase + n) * 64 + cc * 32 + khi);
  v8bf lo = *(const v8bf*)p;
  v8bf hi = *(const v8bf*)(p + 16);
  return __builtin_shufflevector(lo, hi, 0, 1, 2, 3, 4, 5, 6, 7,
                                 8, 9, 10, 11, 12, 13, 14, 15);
}

// ------------------------------ kernels ------------------------------------
__global__ void k_params(const float* __restrict__ nul, const float* __restrict__ thl,
                         const float* __restrict__ dnu, const float* __restrict__ dth,
                         float* __restrict__ lamr, float* __restrict__ lami,
                         float* __restrict__ gam) {
  int i = blockIdx.x * blockDim.x + threadIdx.x;
  if (i >= CR_) return;
  float nu  = expf(nul[i] + dnu[i]);
  float th  = expf(thl[i] + dth[i]);
  float mag = expf(-nu);
  lamr[i] = mag * cosf(th);
  lami[i] = mag * sinf(th);
  gam[i]  = sqrtf(fmaxf(1.f - mag * mag, 1e-6f));
}

// ctx[b,l,c] = mean over (S,W)
__global__ void k_ctx(const float* __restrict__ x, float* __restrict__ ctx) {
  const float* p = x + (size_t)blockIdx.x * SW_;
  float s = 0.f;
  for (int i = threadIdx.x; i < SW_; i += 256) s += p[i];
  __shared__ float red[256];
  red[threadIdx.x] = s;
  __syncthreads();
  for (int st = 128; st > 0; st >>= 1) {
    if (threadIdx.x < st) red[threadIdx.x] += red[threadIdx.x + st];
    __syncthreads();
  }
  if (threadIdx.x == 0) ctx[blockIdx.x] = red[0] * (1.f / SW_);
}

// forcing = scale * complex( tanh(ctx@w1+b1) @ w2 + b2 )
__global__ void k_forcing(const float* __restrict__ ctx, const float* __restrict__ w1,
                          const float* __restrict__ b1, const float* __restrict__ w2,
                          const float* __restrict__ b2, const float* __restrict__ fscale,
                          float* __restrict__ forcr, float* __restrict__ forci) {
  int bl = blockIdx.x;
  __shared__ float cs[C_];
  __shared__ float hid[MH_];
  if (threadIdx.x < C_) cs[threadIdx.x] = ctx[bl * C_ + threadIdx.x];
  __syncthreads();
  if (threadIdx.x < MH_) {
    float a = b1[threadIdx.x];
    for (int k = 0; k < C_; ++k) a += cs[k] * w1[k * MH_ + threadIdx.x];
    hid[threadIdx.x] = tanhf(a);
  }
  __syncthreads();
  float sc = fscale[0];
  for (int j = threadIdx.x; j < CR_ * 2; j += 256) {
    float a = b2[j];
    for (int k = 0; k < MH_; ++k) a += hid[k] * w2[k * CR_ * 2 + j];
    int cr = j >> 1;
    if (j & 1) forci[(size_t)bl * CR_ + cr] = sc * a;
    else       forcr[(size_t)bl * CR_ + cr] = sc * a;
  }
}

// u[b,l,c,r] = gamma * (x projected onto conj(U),conj(V))[r] * (1 + forcing)
__global__ void k_lowrank(const float* __restrict__ x,
                          const float* __restrict__ Ur, const float* __restrict__ Ui,
                          const float* __restrict__ Vr, const float* __restrict__ Vi,
                          const float* __restrict__ gam,
                          const float* __restrict__ frg, const float* __restrict__ fig,
                          float* __restrict__ u_r, float* __restrict__ u_i) {
  __shared__ alignas(16) float vr[W_ * R_], vi[W_ * R_];  // 16 KB each
  __shared__ float uar[R_], uai[R_];
  int blc = blockIdx.x;
  int c = blc % C_, bl = blc / C_;
  stage_async(vr, Vr + (size_t)c * W_ * R_, W_ * R_ * 4);
  stage_async(vi, Vi + (size_t)c * W_ * R_, W_ * R_ * 4);
  if (threadIdx.x < R_) { uar[threadIdx.x] = 0.f; uai[threadIdx.x] = 0.f; }
  wait_async();
  __syncthreads();
  const float* xb = x + (size_t)blc * SW_;
  int wave = threadIdx.x >> 5;
  int m0 = (wave >> 1) * 16, n0 = (wave & 1) * 16;  // 4 S-tiles x 2 R-tiles
  v8f pr = {}, pi = {};
  for (int k0 = 0; k0 < W_; k0 += 4) {
    v2f a  = ldA4(xb + m0 * W_ + k0, W_);
    v2f br = ldB4(vr + k0 * R_ + n0, R_);
    v2f bi = ldB4(vi + k0 * R_ + n0, R_);
    v2f nbi = {-bi.x, -bi.y};          // conj(V): p_i = -x*Vi
    pr = wmma4(a, br, pr);
    pi = wmma4(a, nbi, pi);
  }
  int lane = threadIdx.x & 31;
  int n = lane & 15, rb = (lane >> 4) * 8;
  int r = n0 + n;
  const float* urp = Ur + (size_t)c * S_ * R_;
  const float* uip = Ui + (size_t)c * S_ * R_;
  float sr = 0.f, si = 0.f;
#pragma unroll
  for (int i = 0; i < 8; ++i) {
    int srow = m0 + rb + i;
    float urv = urp[srow * R_ + r], uiv = uip[srow * R_ + r];
    sr += pr[i] * urv + pi[i] * uiv;   // p * conj(U)
    si += pi[i] * urv - pr[i] * uiv;
  }
  atomicAdd(&uar[r], sr);
  atomicAdd(&uai[r], si);
  __syncthreads();
  if (threadIdx.x < R_) {
    int rr = threadIdx.x, crI = c * R_ + rr;
    float g  = gam[crI];
    float fr = 1.f + frg[(size_t)bl * CR_ + crI];
    float fi = fig[(size_t)bl * CR_ + crI];
    float a = uar[rr], b = uai[rr];
    u_r[(size_t)bl * CR_ + crI] = g * (a * fr - b * fi);
    u_i[(size_t)bl * CR_ + crI] = g * (a * fi + b * fr);
  }
}

// diagonal complex recurrence over L (L=16, sequential per (b,c,r))
__global__ void k_scan(const float* __restrict__ u_r, const float* __restrict__ u_i,
                       const float* __restrict__ lamr, const float* __restrict__ lami,
                       float* __restrict__ h_r, float* __restrict__ h_i) {
  int idx = blockIdx.x * blockDim.x + threadIdx.x;  // b*CR + cr
  if (idx >= B_ * CR_) return;
  int b = idx / CR_, cr = idx % CR_;
  float lr = lamr[cr], li = lami[cr];
  float hr = 0.f, hi = 0.f;
  for (int l = 0; l < L_; ++l) {
    size_t o = ((size_t)(b * L_ + l)) * CR_ + cr;
    float nr = lr * hr - li * hi + u_r[o];
    float ni = lr * hi + li * hr + u_i[o];
    hr = nr; hi = ni;
    h_r[o] = hr; h_i[o] = hi;
  }
}

// y[s,w] = sum_r h[r]*U[s,r]*V[w,r]   (f32 WMMA, 4 real GEMMs)
__global__ void k_recon(const float* __restrict__ hrg, const float* __restrict__ hig,
                        const float* __restrict__ Ur, const float* __restrict__ Ui,
                        const float* __restrict__ Vr, const float* __restrict__ Vi,
                        float* __restrict__ y_r, float* __restrict__ y_i) {
  __shared__ float ar[S_ * R_], ai[S_ * R_];              // 8 KB each
  __shared__ alignas(16) float vr[W_ * R_], vi[W_ * R_];  // 16 KB each
  int blc = blockIdx.x;
  int c = blc % C_, bl = blc / C_;
  stage_async(vr, Vr + (size_t)c * W_ * R_, W_ * R_ * 4);
  stage_async(vi, Vi + (size_t)c * W_ * R_, W_ * R_ * 4);
  const float* hrp = hrg + (size_t)bl * CR_ + c * R_;
  const float* hip = hig + (size_t)bl * CR_ + c * R_;
  const float* urp = Ur + (size_t)c * S_ * R_;
  const float* uip = Ui + (size_t)c * S_ * R_;
  for (int i = threadIdx.x; i < S_ * R_; i += 256) {
    int r = i & (R_ - 1);
    float hr = hrp[r], hi = hip[r];
    float u1 = urp[i], u2 = uip[i];
    ar[i] = hr * u1 - hi * u2;          // a = h * U
    ai[i] = hr * u2 + hi * u1;
  }
  wait_async();
  __syncthreads();
  int wave = threadIdx.x >> 5;
  int n0 = wave * 16;                    // 8 W-tiles, one per wave
  for (int mt = 0; mt < 4; ++mt) {
    int m0 = mt * 16;
    v8f yr8 = {}, yi8 = {};
    for (int k0 = 0; k0 < R_; k0 += 4) {
      v2f afr = ldA4(ar + m0 * R_ + k0, R_);
      v2f afi = ldA4(ai + m0 * R_ + k0, R_);
      v2f bfr = ldB4T(vr, n0, k0);
      v2f bfi = ldB4T(vi, n0, k0);
      v2f nfi = {-afi.x, -afi.y};
      yr8 = wmma4(afr, bfr, yr8);        // yr = ar*Vr - ai*Vi
      yr8 = wmma4(nfi, bfi, yr8);
      yi8 = wmma4(afr, bfi, yi8);        // yi = ar*Vi + ai*Vr
      yi8 = wmma4(afi, bfr, yi8);
    }
    stC(y_r + (size_t)blc * SW_ + m0 * W_ + n0, W_, yr8);
    stC(y_i + (size_t)blc * SW_ + m0 * W_ + n0, W_, yi8);
  }
}

// zero the 1-element border of the padded z buffers (channel-innermost layout)
__global__ void k_zpad(__bf16* __restrict__ zpr, __bf16* __restrict__ zpi) {
  size_t base = (size_t)blockIdx.x * SP_ * WP_ * C_;
  const int rowE = WP_ * C_;  // 8320
  __bf16 z = (__bf16)0.f;
  for (int i = threadIdx.x; i < rowE; i += 256) {
    zpr[base + i] = z; zpr[base + (size_t)(SP_ - 1) * rowE + i] = z;
    zpi[base + i] = z; zpi[base + (size_t)(SP_ - 1) * rowE + i] = z;
  }
  for (int t = threadIdx.x; t < (SP_ - 2) * C_; t += 256) {
    int sp = 1 + t / C_, ch = t % C_;
    size_t i0 = base + ((size_t)sp * WP_) * C_ + ch;
    size_t i1 = base + ((size_t)sp * WP_ + WP_ - 1) * C_ + ch;
    zpr[i0] = z; zpr[i1] = z; zpi[i0] = z; zpi[i1] = z;
  }
}

// repack conv weights to bf16 A-fragment layout: wA[(dy*3+dx)][cc][o][kk]
__global__ void k_convw(const float* __restrict__ kr, const float* __restrict__ ki,
                        __bf16* __restrict__ wAr, __bf16* __restrict__ wAi) {
  int j = blockIdx.x * blockDim.x + threadIdx.x;
  if (j >= 9 * 2 * C_ * 32) return;
  int kk = j & 31, o = (j >> 5) & 63, cc = (j >> 11) & 1, off = j >> 12;
  int ci = cc * 32 + kk, dy = off / 3, dx = off % 3;
  int src = ((o * C_ + ci) * 3 + dy) * 3 + dx;
  wAr[j] = (__bf16)kr[src];
  wAi[j] = (__bf16)ki[src];
}

// z = projW * y + projb (complex C x C over N = S*W); bf16 output into padded
// channel-innermost layout.  f32 WMMA, one s-row per workgroup.
__global__ void k_proj(const float* __restrict__ yr, const float* __restrict__ yi,
                       const float* __restrict__ pwr, const float* __restrict__ pwi,
                       const float* __restrict__ pbr, const float* __restrict__ pbi,
                       __bf16* __restrict__ zpr, __bf16* __restrict__ zpi) {
  __shared__ alignas(16) float wr[C_ * C_], wi[C_ * C_];  // 16 KB each
  int bls = blockIdx.x;
  int s = bls % S_, bl = bls / S_;
  stage_async(wr, pwr, C_ * C_ * 4);
  stage_async(wi, pwi, C_ * C_ * 4);
  wait_async();
  __syncthreads();
  int wave = threadIdx.x >> 5;
  int n0 = wave * 16;
  size_t ybase = ((size_t)bl * C_) * SW_ + (size_t)s * W_ + n0;
  for (int mt = 0; mt < 4; ++mt) {
    int m0 = mt * 16;
    v8f zr = {}, zi = {};
    for (int k0 = 0; k0 < C_; k0 += 4) {
      v2f arf = ldA4(wr + m0 * C_ + k0, C_);
      v2f aif = ldA4(wi + m0 * C_ + k0, C_);
      v2f brf = ldB4(yr + ybase + (size_t)k0 * SW_, SW_);
      v2f bif = ldB4(yi + ybase + (size_t)k0 * SW_, SW_);
      v2f nbi = {-bif.x, -bif.y};
      zr = wmma4(arf, brf, zr);          // zr = Wr*yr - Wi*yi
      zr = wmma4(aif, nbi, zr);
      zi = wmma4(arf, bif, zi);          // zi = Wr*yi + Wi*yr
      zi = wmma4(aif, brf, zi);
    }
    int lane = threadIdx.x & 31;
    int n = lane & 15, rb = (lane >> 4) * 8;
    // 8 consecutive channels per lane -> one 16B bf16 store per buffer.
    size_t pbase = (((size_t)bl * SP_ + (s + 1)) * WP_ + (n0 + n + 1)) * C_
                   + m0 + rb;
    v8bf pr8, pi8;
#pragma unroll
    for (int i = 0; i < 8; ++i) {
      int o = m0 + rb + i;
      pr8[i] = (__bf16)(zr[i] + pbr[o]);
      pi8[i] = (__bf16)(zi[i] + pbi[o]);
    }
    *(v8bf*)(zpr + pbase) = pr8;
    *(v8bf*)(zpi + pbase) = pi8;
  }
}

// 3x3 conv (C=64 -> 64) as implicit GEMM with bf16 16x16x32 WMMA.
// One (b,l,s) row per workgroup; halo slab [3 rows][130 w][64 ch] staged with
// async b128 copies (fully contiguous source in channel-innermost layout).
__global__ void k_conv(const __bf16* __restrict__ zp, const __bf16* __restrict__ wA,
                       const float* __restrict__ bias, float* __restrict__ out) {
  __shared__ alignas(16) __bf16 slab[3 * WP_ * C_];  // 49,920 B
  int bls = blockIdx.x;
  int s = bls % S_, bl = bls / S_;
  if (threadIdx.x < 144)  // prefetch full weight block (73,728 B)
    __builtin_prefetch((const char*)wA + threadIdx.x * 512, 0, 0);
  stage_async(slab, zp + (((size_t)bl * SP_ + s) * WP_) * C_, 3 * WP_ * C_ * 2);
  wait_async();
  __syncthreads();
  int wave = threadIdx.x >> 5;
  int n0 = wave * 16;
  for (int mt = 0; mt < 4; ++mt) {
    int m0 = mt * 16;
    v8f acc = {};
    for (int off = 0; off < 9; ++off) {
      int dy = off / 3, dx = off % 3;
#pragma unroll
      for (int cc = 0; cc < 2; ++cc) {
        v16bf a = ldA16v(wA + ((off * 2 + cc) * C_ + m0) * 32, 32);
        v16bf b = ldB16slab(slab, cc, dy, n0 + dx);
        acc = wmma32bf(a, b, acc);
      }
    }
    int lane = threadIdx.x & 31;
    int n = lane & 15, rb = (lane >> 4) * 8;
#pragma unroll
    for (int i = 0; i < 8; ++i) {
      int o = m0 + rb + i;
      out[((size_t)(bl * C_ + o)) * SW_ + (size_t)s * W_ + n0 + n] = acc[i] + bias[o];
    }
  }
}

// 1x1 fuse: fused[o] = fuse_b[o] + sum_c fk[o][c]*cr[c] + fk[o][64+c]*ci[c]
__global__ void k_fuse(const float* __restrict__ cr, const float* __restrict__ ci,
                       const float* __restrict__ fk, const float* __restrict__ fb,
                       float* __restrict__ fused) {
  __shared__ alignas(16) float wf[C_ * 2 * C_];  // 32 KB
  int bls = blockIdx.x;
  int s = bls % S_, bl = bls / S_;
  stage_async(wf, fk, C_ * 2 * C_ * 4);
  wait_async();
  __syncthreads();
  int wave = threadIdx.x >> 5;
  int n0 = wave * 16;
  for (int mt = 0; mt < 4; ++mt) {
    int m0 = mt * 16;
    v8f acc = {};
    const float* bb = cr + ((size_t)bl * C_) * SW_ + (size_t)s * W_ + n0;
    for (int k0 = 0; k0 < C_; k0 += 4) {
      v2f a = ldA4(wf + m0 * (2 * C_) + k0, 2 * C_);
      v2f b = ldB4(bb + (size_t)k0 * SW_, SW_);
      acc = wmma4(a, b, acc);
    }
    bb = ci + ((size_t)bl * C_) * SW_ + (size_t)s * W_ + n0;
    for (int k0 = 0; k0 < C_; k0 += 4) {
      v2f a = ldA4(wf + m0 * (2 * C_) + C_ + k0, 2 * C_);
      v2f b = ldB4(bb + (size_t)k0 * SW_, SW_);
      acc = wmma4(a, b, acc);
    }
    int lane = threadIdx.x & 31;
    int n = lane & 15, rb = (lane >> 4) * 8;
#pragma unroll
    for (int i = 0; i < 8; ++i) {
      int o = m0 + rb + i;
      fused[((size_t)(bl * C_ + o)) * SW_ + (size_t)s * W_ + n0 + n] = acc[i] + fb[o];
    }
  }
}

// LayerNorm over (S,W) per (b,l,c), affine (ln_g, ln_b), residual +x.
__global__ void k_ln(const float* __restrict__ fused, const float* __restrict__ x,
                     const float* __restrict__ g, const float* __restrict__ bta,
                     float* __restrict__ out) {
  size_t base = (size_t)blockIdx.x * SW_;
  float s1 = 0.f, s2 = 0.f;
  for (int i = threadIdx.x; i < SW_; i += 256) {
    float v = fused[base + i];
    s1 += v; s2 += v * v;
  }
  __shared__ float r1[256], r2[256];
  r1[threadIdx.x] = s1; r2[threadIdx.x] = s2;
  __syncthreads();
  for (int st = 128; st > 0; st >>= 1) {
    if (threadIdx.x < st) { r1[threadIdx.x] += r1[threadIdx.x + st];
                            r2[threadIdx.x] += r2[threadIdx.x + st]; }
    __syncthreads();
  }
  float mu   = r1[0] * (1.f / SW_);
  float var  = r2[0] * (1.f / SW_) - mu * mu;
  float rstd = rsqrtf(var + 1e-5f);
  for (int i = threadIdx.x; i < SW_; i += 256) {
    out[base + i] = (fused[base + i] - mu) * rstd * g[i] + bta[i] + x[base + i];
  }
}

// ------------------------------ launcher -----------------------------------
extern "C" void kernel_launch(void* const* d_in, const int* in_sizes, int n_in,
                              void* d_out, int out_size, void* d_ws, size_t ws_size,
                              hipStream_t stream) {
  const float* x      = (const float*)d_in[0];
  const float* nul    = (const float*)d_in[1];
  const float* thl    = (const float*)d_in[2];
  const float* dnu    = (const float*)d_in[3];
  const float* dth    = (const float*)d_in[4];
  const float* w1     = (const float*)d_in[5];
  const float* b1     = (const float*)d_in[6];
  const float* w2     = (const float*)d_in[7];
  const float* b2     = (const float*)d_in[8];
  const float* fscale = (const float*)d_in[9];
  const float* Ur     = (const float*)d_in[10];
  const float* Ui     = (const float*)d_in[11];
  const float* Vr     = (const float*)d_in[12];
  const float* Vi     = (const float*)d_in[13];
  const float* pwr    = (const float*)d_in[14];
  const float* pwi    = (const float*)d_in[15];
  const float* pbr    = (const float*)d_in[16];
  const float* pbi    = (const float*)d_in[17];
  const float* convrk = (const float*)d_in[18];
  const float* convrb = (const float*)d_in[19];
  const float* convik = (const float*)d_in[20];
  const float* convib = (const float*)d_in[21];
  const float* fusek  = (const float*)d_in[22];
  const float* fuseb  = (const float*)d_in[23];
  const float* lng    = (const float*)d_in[24];
  const float* lnb    = (const float*)d_in[25];
  float* out = (float*)d_out;

  char* wsb = (char*)d_ws;
  size_t off = 0;
  auto alloc = [&](size_t bytes) -> char* {
    char* p = wsb + off;
    off = (off + bytes + 255) & ~(size_t)255;
    return p;
  };
  float*  ctx   = (float*)alloc((size_t)BL_ * C_ * 4);
  float*  lamr  = (float*)alloc((size_t)CR_ * 4);
  float*  lami  = (float*)alloc((size_t)CR_ * 4);
  float*  gam   = (float*)alloc((size_t)CR_ * 4);
  float*  forcr = (float*)alloc((size_t)BL_ * CR_ * 4);
  float*  forci = (float*)alloc((size_t)BL_ * CR_ * 4);
  float*  u_r   = (float*)alloc((size_t)BL_ * CR_ * 4);
  float*  u_i   = (float*)alloc((size_t)BL_ * CR_ * 4);
  float*  h_r   = (float*)alloc((size_t)BL_ * CR_ * 4);
  float*  h_i   = (float*)alloc((size_t)BL_ * CR_ * 4);
  __bf16* wAr   = (__bf16*)alloc((size_t)9 * 2 * C_ * 32 * 2);
  __bf16* wAi   = (__bf16*)alloc((size_t)9 * 2 * C_ * 32 * 2);
  size_t zp_elems = (size_t)BL_ * C_ * SP_ * WP_;       // 17,571,840
  __bf16* zpr = (__bf16*)alloc(zp_elems * 2);
  __bf16* zpi = (__bf16*)alloc(zp_elems * 2);
  float* fused = (float*)zpr;  // alias: 67.1MB fused fits in 70.3MB zp region
  size_t y_elems = (size_t)BL_ * C_ * SW_;              // 16,777,216
  float* y_r = (float*)alloc(y_elems * 4);
  float* y_i = (float*)alloc(y_elems * 4);
  (void)in_sizes; (void)n_in; (void)out_size; (void)ws_size;

  dim3 T(256);
  k_params <<<dim3((CR_ + 255) / 256), T, 0, stream>>>(nul, thl, dnu, dth, lamr, lami, gam);
  k_ctx    <<<dim3(BL_ * C_),          T, 0, stream>>>(x, ctx);
  k_forcing<<<dim3(BL_),               T, 0, stream>>>(ctx, w1, b1, w2, b2, fscale, forcr, forci);
  k_lowrank<<<dim3(BL_ * C_),          T, 0, stream>>>(x, Ur, Ui, Vr, Vi, gam, forcr, forci, u_r, u_i);
  k_scan   <<<dim3((B_ * CR_ + 255) / 256), T, 0, stream>>>(u_r, u_i, lamr, lami, h_r, h_i);
  k_recon  <<<dim3(BL_ * C_),          T, 0, stream>>>(h_r, h_i, Ur, Ui, Vr, Vi, y_r, y_i);
  k_zpad   <<<dim3(BL_),               T, 0, stream>>>(zpr, zpi);
  k_convw  <<<dim3((9 * 2 * C_ * 32 + 255) / 256), T, 0, stream>>>(convrk, convik, wAr, wAi);
  k_proj   <<<dim3(BL_ * S_),          T, 0, stream>>>(y_r, y_i, pwr, pwi, pbr, pbi, zpr, zpi);
  k_conv   <<<dim3(BL_ * S_),          T, 0, stream>>>(zpr, wAr, convrb, y_r);
  k_conv   <<<dim3(BL_ * S_),          T, 0, stream>>>(zpi, wAi, convib, y_i);
  k_fuse   <<<dim3(BL_ * S_),          T, 0, stream>>>(y_r, y_i, fusek, fuseb, fused);
  k_ln     <<<dim3(BL_ * C_),          T, 0, stream>>>(fused, x, lng, lnb, out);
}
